// Graph_69483980914792
// MI455X (gfx1250) — matrix operand order repaired and checked by reading, not compiled, not run on hardware
//
#include <hip/hip_runtime.h>

typedef __attribute__((ext_vector_type(16))) __bf16 v16bf;
typedef __attribute__((ext_vector_type(8)))  __bf16 v8bf;
typedef __attribute__((ext_vector_type(8)))  float  v8f;

#define NODES   20000
#define NREL    8
#define EDGES   320000
#define IN_DIM  1280
#define HID_DIM 512
#define OUT_DIM 256

// ---------------------------------------------------------------- utilities

__global__ void zero_i32(int* __restrict__ p, int n) {
  int i = blockIdx.x * blockDim.x + threadIdx.x;
  if (i < n) p[i] = 0;
}

// per-(dst, relation) in-degree, shared by both layers (reference semantics)
__global__ void count_deg(const int* __restrict__ dst, const int* __restrict__ et,
                          int* __restrict__ cnt) {
  int e = blockIdx.x * blockDim.x + threadIdx.x;
  if (e < EDGES) atomicAdd(&cnt[dst[e] * NREL + et[e]], 1);
}

__global__ void cast_bf16(const float* __restrict__ in, __bf16* __restrict__ out, int n) {
  int i = blockIdx.x * blockDim.x + threadIdx.x;
  if (i < n) out[i] = (__bf16)in[i];
}

// Repack R+1 weight matrices ([K x N] row-major fp32; r < NREL from W, r == NREL
// from root) into per-lane v_wmma_f32_16x16x32_bf16 B-fragment order:
//   element (r, nt, kb, lane, j) <- B[r][kb*32 + (lane>>4)*8 + (j&7) + (j>>3)*16][nt*16 + (lane&15)]
// so the GEMM loads one contiguous v16bf (32B) per lane per fragment.
__global__ void repack_weights(const float* __restrict__ W, const float* __restrict__ root,
                               __bf16* __restrict__ out, int K, int N, int total) {
  int p = blockIdx.x * blockDim.x + threadIdx.x;
  if (p >= total) return;
  int KB = K >> 5, NT = N >> 4;
  int j = p & 15;
  int q = p >> 4;
  int lane = q & 31; q >>= 5;
  int kb = q % KB;   q /= KB;
  int nt = q % NT;
  int r  = q / NT;
  int k = (kb << 5) + ((lane >> 4) << 3) + (j & 7) + ((j >> 3) << 4);
  int n = (nt << 4) + (lane & 15);
  float v = (r < NREL) ? W[((size_t)r * K + k) * N + n] : root[(size_t)k * N + n];
  out[p] = (__bf16)v;
}

// ---------------------------------------------------------------- WMMA GEMM
// A: [M x K] bf16 row-major. Bp: packed fragments for NREL+1 matrices.
// One wave computes a 32x64 output strip: 2 m-tiles x 4 n-tiles, 8 accumulators.
// Each B fragment feeds 2 WMMAs, each A fragment 4 WMMAs (2x the arithmetic
// intensity of a 16x64 strip). Operands stream from the 192MB L2.
// r < NREL  -> store bf16 into Hout[r]   (per-relation transform)
// r == NREL -> store fp32 root product + bias into Rout (aggregation base)
__global__ __launch_bounds__(256) void rgcn_gemm(const __bf16* __restrict__ A,
                                                 const __bf16* __restrict__ Bp,
                                                 __bf16* __restrict__ Hout,
                                                 float* __restrict__ Rout,
                                                 const float* __restrict__ bias,
                                                 int M, int K, int N) {
  const int ntg        = N >> 6;          // groups of four 16-wide n-tiles
  const int jobs_per_r = (M >> 5) * ntg;  // 32-row m-strips
  const int wave = blockIdx.x * (blockDim.x >> 5) + (threadIdx.x >> 5);
  if (wave >= (NREL + 1) * jobs_per_r) return;
  const int lane = threadIdx.x & 31;
  const int r   = wave / jobs_per_r;
  const int rem = wave - r * jobs_per_r;
  const int mt  = rem / ntg;              // 32-row strip index
  const int ng  = rem - mt * ntg;

  const int g  = lane >> 4;
  const int ml = lane & 15;
  const int KB = K >> 5;
  const int NT = N >> 4;

  // A-fragment lane layout (16-bit 16x32): K = (j&7) + (j>>3)*16 + g*8
  // -> two contiguous 16B runs per lane: [g*8, g*8+7] and [16+g*8, 16+g*8+7]
  const __bf16* arow0 = A + (size_t)(mt * 32 + ml) * K + g * 8;
  const __bf16* arow1 = arow0 + (size_t)16 * K;
  const __bf16* bbase = Bp + (((size_t)((size_t)r * NT + ng * 4) * KB) * 32 + lane) * 16;
  const size_t  bnt   = (size_t)KB * 512;   // elements between adjacent n-tiles

  v8f acc[8];
#pragma unroll
  for (int t = 0; t < 8; ++t)
#pragma unroll
    for (int i = 0; i < 8; ++i) acc[t][i] = 0.0f;

  for (int kb = 0; kb < KB; ++kb) {
    v8bf a0lo = *(const v8bf*)(arow0 + kb * 32);
    v8bf a0hi = *(const v8bf*)(arow0 + kb * 32 + 16);
    v8bf a1lo = *(const v8bf*)(arow1 + kb * 32);
    v8bf a1hi = *(const v8bf*)(arow1 + kb * 32 + 16);
    v16bf a0, a1;
#pragma unroll
    for (int i = 0; i < 8; ++i) {
      a0[i] = a0lo[i]; a0[i + 8] = a0hi[i];
      a1[i] = a1lo[i]; a1[i + 8] = a1hi[i];
    }
    const __bf16* bk = bbase + (size_t)kb * 512;
#pragma unroll
    for (int t = 0; t < 4; ++t) {
      v16bf b = *(const v16bf*)(bk + (size_t)t * bnt);
      acc[t]     = __builtin_amdgcn_wmma_f32_16x16x32_bf16(false, a0, false, b, (short)0,
                                                           acc[t],     false, false);
      acc[t + 4] = __builtin_amdgcn_wmma_f32_16x16x32_bf16(false, a1, false, b, (short)0,
                                                           acc[t + 4], false, false);
    }
  }

  // C/D layout: VGPR v -> row (v + g*8), lane&15 -> column
  if (r < NREL) {
    __bf16* o = Hout + ((size_t)r * M + (size_t)mt * 32) * N + ng * 64;
#pragma unroll
    for (int u = 0; u < 2; ++u)
#pragma unroll
      for (int t = 0; t < 4; ++t)
#pragma unroll
        for (int v = 0; v < 8; ++v)
          o[(size_t)(u * 16 + v + g * 8) * N + (t * 16 + ml)] = (__bf16)acc[u * 4 + t][v];
  } else {
    float* o = Rout + (size_t)(mt * 32) * N + ng * 64;
#pragma unroll
    for (int u = 0; u < 2; ++u)
#pragma unroll
      for (int t = 0; t < 4; ++t)
#pragma unroll
        for (int v = 0; v < 8; ++v) {
          int nn = t * 16 + ml;
          o[(size_t)(u * 16 + v + g * 8) * N + nn] = acc[u * 4 + t][v] + bias[ng * 64 + nn];
        }
  }
}

// ------------------------------------------------------------ edge scatter
// msg = H[etype][src] / deg(dst, etype); atomically accumulated into out[dst].
// dim/8 threads per edge; each thread moves 8 bf16 (one 16B load).
__global__ void edge_aggregate(const int* __restrict__ src, const int* __restrict__ dst,
                               const int* __restrict__ et, const int* __restrict__ cnt,
                               const __bf16* __restrict__ H, float* __restrict__ out,
                               int dim, int lg) {
  int tid = blockIdx.x * blockDim.x + threadIdx.x;
  int e = tid >> lg;
  if (e >= EDGES) return;
  int c = (tid & ((1 << lg) - 1)) << 3;
  int d = dst[e], s = src[e], t = et[e];
  int deg = cnt[d * NREL + t];
  float inv = 1.0f / (float)(deg > 0 ? deg : 1);
  v8bf m = *(const v8bf*)(H + ((size_t)t * NODES + s) * dim + c);
  float* o = out + (size_t)d * dim + c;
#pragma unroll
  for (int i = 0; i < 8; ++i) atomicAdd(&o[i], (float)m[i] * inv);
}

__global__ void relu_cast(const float* __restrict__ h, __bf16* __restrict__ hb, int n) {
  int i = blockIdx.x * blockDim.x + threadIdx.x;
  if (i < n) {
    float v = h[i];
    hb[i] = (__bf16)(v > 0.0f ? v : 0.0f);
  }
}

// ---------------------------------------------------------------- launcher

extern "C" void kernel_launch(void* const* d_in, const int* in_sizes, int n_in,
                              void* d_out, int out_size, void* d_ws, size_t ws_size,
                              hipStream_t stream) {
  (void)in_sizes; (void)n_in; (void)out_size; (void)ws_size;
  const float* x     = (const float*)d_in[0];
  const int*   ei    = (const int*)d_in[1];
  const int*   etype = (const int*)d_in[2];
  const float* W1    = (const float*)d_in[3];
  const float* root1 = (const float*)d_in[4];
  const float* b1    = (const float*)d_in[5];
  const float* W2    = (const float*)d_in[6];
  const float* root2 = (const float*)d_in[7];
  const float* b2    = (const float*)d_in[8];
  const int* src = ei;                 // edge_index[0]
  const int* dst = ei + EDGES;         // edge_index[1]
  float* out = (float*)d_out;

  char* ws = (char*)d_ws;
  size_t off = 0;
  auto carve = [&](size_t bytes) -> char* {
    char* p = ws + off;
    off += (bytes + 255) & ~(size_t)255;
    return p;
  };
  int*    cnt = (int*)   carve((size_t)NODES * NREL * sizeof(int));       // 0.64 MB
  __bf16* xb  = (__bf16*)carve((size_t)NODES * IN_DIM * 2);               // 51.2 MB
  __bf16* w1p = (__bf16*)carve((size_t)(NREL + 1) * IN_DIM * HID_DIM * 2);// 11.8 MB
  __bf16* w2p = (__bf16*)carve((size_t)(NREL + 1) * HID_DIM * OUT_DIM * 2);// 2.4 MB
  __bf16* H1  = (__bf16*)carve((size_t)NREL * NODES * HID_DIM * 2);       // 163.8 MB
  float*  h   = (float*) carve((size_t)NODES * HID_DIM * sizeof(float));  // 41.0 MB
  __bf16* hb  = (__bf16*)carve((size_t)NODES * HID_DIM * 2);              // 20.5 MB
  __bf16* H2  = (__bf16*)carve((size_t)NREL * NODES * OUT_DIM * 2);       // 81.9 MB

  // degrees (shared by both layers)
  zero_i32<<<(NODES * NREL + 255) / 256, 256, 0, stream>>>(cnt, NODES * NREL);
  count_deg<<<(EDGES + 255) / 256, 256, 0, stream>>>(dst, etype, cnt);

  // bf16 casts + weight fragment pre-packing
  cast_bf16<<<(NODES * IN_DIM + 255) / 256, 256, 0, stream>>>(x, xb, NODES * IN_DIM);
  int w1tot = (NREL + 1) * IN_DIM * HID_DIM;
  repack_weights<<<(w1tot + 255) / 256, 256, 0, stream>>>(W1, root1, w1p, IN_DIM, HID_DIM, w1tot);
  int w2tot = (NREL + 1) * HID_DIM * OUT_DIM;
  repack_weights<<<(w2tot + 255) / 256, 256, 0, stream>>>(W2, root2, w2p, HID_DIM, OUT_DIM, w2tot);

  // ---- layer 1: H1[r] = x @ W1[r] (bf16), h = x @ root1 + b1 (fp32)
  int jobs1 = (NREL + 1) * (NODES / 32) * (HID_DIM / 64);   // 45000 waves
  rgcn_gemm<<<(jobs1 + 7) / 8, 256, 0, stream>>>(xb, w1p, H1, h, b1, NODES, IN_DIM, HID_DIM);
  edge_aggregate<<<(EDGES * (HID_DIM / 8) + 255) / 256, 256, 0, stream>>>(
      src, dst, etype, cnt, H1, h, HID_DIM, 6);
  relu_cast<<<(NODES * HID_DIM + 255) / 256, 256, 0, stream>>>(h, hb, NODES * HID_DIM);

  // ---- layer 2: H2[r] = relu(h) @ W2[r] (bf16), out = relu(h) @ root2 + b2
  int jobs2 = (NREL + 1) * (NODES / 32) * (OUT_DIM / 64);   // 22500 waves
  rgcn_gemm<<<(jobs2 + 7) / 8, 256, 0, stream>>>(hb, w2p, H2, out, b2, NODES, HID_DIM, OUT_DIM);
  edge_aggregate<<<(EDGES * (OUT_DIM / 8) + 255) / 256, 256, 0, stream>>>(
      src, dst, etype, cnt, H2, out, OUT_DIM, 5);
}
